// MSDeformAttnWrapper_52140902974274
// MI455X (gfx1250) — matrix operand order repaired and checked by reading, not compiled, not run on hardware
//
#include <hip/hip_runtime.h>
#include <hip/hip_fp16.h>
#include <math.h>

#define D_MODEL  128
#define N_HEADS  16
#define N_LEVELS 3
#define N_POINTS 4
#define D_HEAD   8
#define LQ       5376      // 64*64 + 32*32 + 16*16
#define BATCH    8
#define N_OA     576       // 384 offset cols + 192 attn cols
#define KT       4         // 128 / 32 k-tiles per GEMM

typedef __attribute__((ext_vector_type(16))) _Float16 v16h;
typedef __attribute__((ext_vector_type(8)))  _Float16 v8h;
typedef __attribute__((ext_vector_type(8)))  float    v8f;

// ---------------------------------------------------------------------------
// Level metadata (H == W per level in this problem)
// ---------------------------------------------------------------------------
__device__ __forceinline__ void level_info(int q, int& lvl, int& start, int& Wl) {
    if (q < 4096)      { lvl = 0; start = 0;    Wl = 64; }
    else if (q < 5120) { lvl = 1; start = 4096; Wl = 32; }
    else               { lvl = 2; start = 5120; Wl = 16; }
}

// ---------------------------------------------------------------------------
// WMMA fragment loaders (wave32, 16x16x32 f16 -> f32)
//   A 16x32: lane L holds row (L&15); halfs 0..7 = K kbase+(L<16?0:8)+i,
//            halfs 8..15 = K kbase+16+(L<16?0:8)+i  (ISA 05_wmma 7.12.2)
//   B 32x16: pre-swizzled in memory so lane L reads 16 contiguous halfs.
// ---------------------------------------------------------------------------
__device__ __forceinline__ v16h load_a_frag(const _Float16* __restrict__ xrow,
                                            int lane, int kbase) {
    int m  = lane & 15;
    int k0 = kbase + ((lane & 16) ? 8 : 0);
    const _Float16* p = xrow + (size_t)m * D_MODEL + k0;
    v8h lo = *(const v8h*)(p);
    v8h hi = *(const v8h*)(p + 16);
    v16h a;
#pragma unroll
    for (int i = 0; i < 8; ++i) { a[i] = lo[i]; a[8 + i] = hi[i]; }
    return a;
}

__device__ __forceinline__ v8f wmma_tile(const v16h a[KT],
                                         const _Float16* __restrict__ frag,
                                         int nt, int NT, int lane) {
    v8f acc = {};
#pragma unroll
    for (int kt = 0; kt < KT; ++kt) {
        v16h b = *(const v16h*)(frag + ((size_t)(kt * NT + nt) * 512) + lane * 16);
        acc = __builtin_amdgcn_wmma_f32_16x16x32_f16(
            /*neg_a=*/false, a[kt], /*neg_b=*/false, b,
            /*c_mod=*/(short)0, acc, /*reuse_a=*/false, /*reuse_b=*/false);
    }
    return acc;
}

// ---------------------------------------------------------------------------
// Kernel 1: pack feature maps (B,128,H,W) -> token-major fp16 x (B*Lq,128)
// LDS-tiled transpose: coalesced f32 reads, coalesced fp16 row writes.
// Block = 256 threads handles a 128(channel) x 32(spatial) tile.
// ---------------------------------------------------------------------------
__global__ void pack_x_kernel(const float* __restrict__ fm1,
                              const float* __restrict__ fm2,
                              const float* __restrict__ fm3,
                              _Float16* __restrict__ xh) {
    __shared__ float tile[D_MODEL * 33];    // padded stride: conflict-free
    int bx = blockIdx.x;                    // 0..167  (s-block within batch)
    int b  = blockIdx.y;
    const float* fm; int start, hw, s0;
    if (bx < 128)      { fm = fm1; start = 0;    hw = 4096; s0 = bx * 32; }
    else if (bx < 160) { fm = fm2; start = 4096; hw = 1024; s0 = (bx - 128) * 32; }
    else               { fm = fm3; start = 5120; hw = 256;  s0 = (bx - 160) * 32; }
    int tid = threadIdx.x;

    // phase 1: read (c, s0..s0+31) coalesced along s
#pragma unroll
    for (int i = 0; i < 16; ++i) {
        int idx = i * 256 + tid;            // 0..4095
        int c = idx >> 5;
        int s = idx & 31;
        tile[c * 33 + s] = fm[((size_t)b * D_MODEL + c) * hw + s0 + s];
    }
    __syncthreads();

    // phase 2: write token-major rows, contiguous 256B per token
#pragma unroll
    for (int i = 0; i < 16; ++i) {
        int idx = i * 256 + tid;
        int s = idx >> 7;                   // 0..31
        int c = idx & 127;
        xh[((size_t)b * LQ + start + s0 + s) * D_MODEL + c] =
            (_Float16)tile[c * 33 + s];
    }
}

// ---------------------------------------------------------------------------
// Kernel 2: swizzle all weights (fp32 row-major K x N) into fp16 B-fragments.
// Fragment flat index t: e = t&15, lane = (t>>4)&31, rem = t>>9,
//   nt = rem % (N/16), kt = rem / (N/16)
//   source k = kt*32 + (e&8 ? 16 : 0) + (lane&16 ? 8 : 0) + (e&7) ; n = nt*16 + (lane&15)
// ---------------------------------------------------------------------------
__global__ void pack_w_kernel(const float* __restrict__ W_val,
                              const float* __restrict__ W_off,
                              const float* __restrict__ W_attn,
                              const float* __restrict__ W_out,
                              _Float16* __restrict__ wval,
                              _Float16* __restrict__ woff,
                              _Float16* __restrict__ wattn,
                              _Float16* __restrict__ wout) {
    const int SZ_VAL = 128 * 128, SZ_OFF = 128 * 384, SZ_ATTN = 128 * 192, SZ_OUT = 128 * 128;
    int tid = blockIdx.x * blockDim.x + threadIdx.x;
    const float* src; _Float16* dst; int N; int t;
    if (tid < SZ_VAL)                          { src = W_val;  dst = wval;  N = 128; t = tid; }
    else if (tid < SZ_VAL + SZ_OFF)            { src = W_off;  dst = woff;  N = 384; t = tid - SZ_VAL; }
    else if (tid < SZ_VAL + SZ_OFF + SZ_ATTN)  { src = W_attn; dst = wattn; N = 192; t = tid - SZ_VAL - SZ_OFF; }
    else if (tid < SZ_VAL + SZ_OFF + SZ_ATTN + SZ_OUT)
                                               { src = W_out;  dst = wout;  N = 128; t = tid - SZ_VAL - SZ_OFF - SZ_ATTN; }
    else return;
    int e    = t & 15;
    int lane = (t >> 4) & 31;
    int rem  = t >> 9;
    int NT   = N / 16;
    int nt   = rem % NT;
    int kt   = rem / NT;
    int k = kt * 32 + ((e & 8) ? 16 : 0) + ((lane & 16) ? 8 : 0) + (e & 7);
    int n = nt * 16 + (lane & 15);
    dst[t] = (_Float16)src[k * N + n];
}

// ---------------------------------------------------------------------------
// Kernel 3: value = x @ W_val + b_val (fp16 WMMA, fp32 acc), stored as fp16
// token-major (B*Lq,128) to halve the random-gather payload of kernel 4.
// Block = 128 threads (4 wave32), one 16-query tile; wave w -> n-tiles 2w,2w+1
// ---------------------------------------------------------------------------
__global__ void gemm_value_kernel(const _Float16* __restrict__ xh,
                                  const _Float16* __restrict__ wval_frag,
                                  const float* __restrict__ b_val,
                                  _Float16* __restrict__ value) {
    int qt   = blockIdx.x;
    int b    = blockIdx.y;
    int wave = threadIdx.x >> 5;
    int lane = threadIdx.x & 31;
    size_t rowbase = (size_t)b * LQ + (size_t)qt * 16;
    const _Float16* xrow = xh + rowbase * D_MODEL;
    v16h a[KT];
#pragma unroll
    for (int kt = 0; kt < KT; ++kt) a[kt] = load_a_frag(xrow, lane, kt * 32);
    const int NT = D_MODEL / 16;   // 8
#pragma unroll
    for (int i = 0; i < 2; ++i) {
        int nt = wave * 2 + i;
        v8f acc = wmma_tile(a, wval_frag, nt, NT, lane);
        int n = nt * 16 + (lane & 15);
        float bias = b_val[n];
#pragma unroll
        for (int r = 0; r < 8; ++r) {
            int m = r + ((lane & 16) ? 8 : 0);
            value[(rowbase + m) * D_MODEL + n] = (_Float16)(acc[r] + bias);
        }
    }
}

// ---------------------------------------------------------------------------
// Kernel 4 (fused): P = x_tile @ [W_off | W_attn] + bias  (WMMA -> LDS),
// then per (query, head): softmax over 12 logits + bilinear gather from the
// fp16 value tensor (one 16B v8h per corner, L2-resident), fp32 accumulate,
// fp16 sampled tokens out. Block = 128 threads, LDS 16 x 576 f32 = 36 KB.
// ---------------------------------------------------------------------------
__global__ void msda_sample_kernel(const _Float16* __restrict__ xh,
                                   const _Float16* __restrict__ woff_frag,
                                   const _Float16* __restrict__ wattn_frag,
                                   const float* __restrict__ b_off,
                                   const float* __restrict__ b_attn,
                                   const _Float16* __restrict__ value,
                                   _Float16* __restrict__ sampled_h) {
    __shared__ float P[16 * N_OA];
    int qt   = blockIdx.x;
    int b    = blockIdx.y;
    int wave = threadIdx.x >> 5;
    int lane = threadIdx.x & 31;
    size_t rowbase = (size_t)b * LQ + (size_t)qt * 16;
    const _Float16* xrow = xh + rowbase * D_MODEL;

    v16h a[KT];
#pragma unroll
    for (int kt = 0; kt < KT; ++kt) a[kt] = load_a_frag(xrow, lane, kt * 32);

    // 36 n-tiles total (24 offset + 12 attn); 9 per wave
#pragma unroll
    for (int i = 0; i < 9; ++i) {
        int nt = wave * 9 + i;
        v8f acc = (nt < 24) ? wmma_tile(a, woff_frag,  nt,      24, lane)
                            : wmma_tile(a, wattn_frag, nt - 24, 12, lane);
        int col    = nt * 16 + (lane & 15);
        float bias = (col < 384) ? b_off[col] : b_attn[col - 384];
#pragma unroll
        for (int r = 0; r < 8; ++r) {
            int m = r + ((lane & 16) ? 8 : 0);
            P[m * N_OA + col] = acc[r] + bias;
        }
    }
    __syncthreads();

    int qbase = qt * 16;
    for (int rep = 0; rep < 2; ++rep) {
        int task = rep * 128 + threadIdx.x;     // 256 (query,head) tasks
        int ql   = task >> 4;
        int head = task & 15;
        int q    = qbase + ql;
        int lvq, startq, Wq;
        level_info(q, lvq, startq, Wq);
        int s = q - startq;
        float refx = ((s % Wq) + 0.5f) / (float)Wq;
        float refy = ((s / Wq) + 0.5f) / (float)Wq;   // H == W
        const float* Prow = P + ql * N_OA;

        // softmax over 12 attention logits
        float logits[N_LEVELS * N_POINTS];
        float mx = -INFINITY;
#pragma unroll
        for (int j = 0; j < 12; ++j) {
            logits[j] = Prow[384 + head * 12 + j];
            mx = fmaxf(mx, logits[j]);
        }
        float denom = 0.f;
#pragma unroll
        for (int j = 0; j < 12; ++j) { logits[j] = __expf(logits[j] - mx); denom += logits[j]; }
        float inv = 1.f / denom;

        float out[D_HEAD];
#pragma unroll
        for (int d = 0; d < D_HEAD; ++d) out[d] = 0.f;

#pragma unroll
        for (int l = 0; l < N_LEVELS; ++l) {
            int Wl = (l == 0) ? 64 : (l == 1) ? 32 : 16;
            int st = (l == 0) ? 0  : (l == 1) ? 4096 : 5120;
            float invW = 1.f / (float)Wl;
#pragma unroll
            for (int p = 0; p < N_POINTS; ++p) {
                float aw = logits[l * 4 + p] * inv;
                float ox = Prow[head * 24 + l * 8 + p * 2 + 0];
                float oy = Prow[head * 24 + l * 8 + p * 2 + 1];
                float lx = (refx + ox * invW) * (float)Wl - 0.5f;
                float ly = (refy + oy * invW) * (float)Wl - 0.5f;
                float x0 = floorf(lx), y0 = floorf(ly);
                int x0i = (int)x0, y0i = (int)y0;
                float fx = lx - x0, fy = ly - y0;
#pragma unroll
                for (int dy = 0; dy < 2; ++dy) {
#pragma unroll
                    for (int dx = 0; dx < 2; ++dx) {
                        int xi = x0i + dx, yi = y0i + dy;
                        if (xi < 0 || xi >= Wl || yi < 0 || yi >= Wl) continue;
                        float w = (dx ? fx : 1.f - fx) * (dy ? fy : 1.f - fy) * aw;
                        const _Float16* vrow = value
                            + ((size_t)b * LQ + st + yi * Wl + xi) * D_MODEL
                            + head * D_HEAD;
                        v8h v = *(const v8h*)vrow;   // one 16B gather per corner
#pragma unroll
                        for (int d = 0; d < D_HEAD; ++d) out[d] += (float)v[d] * w;
                    }
                }
            }
        }
        v8h res;
#pragma unroll
        for (int d = 0; d < D_HEAD; ++d) res[d] = (_Float16)out[d];
        *(v8h*)(sampled_h + (rowbase + ql) * D_MODEL + head * D_HEAD) = res;
    }
}

// ---------------------------------------------------------------------------
// Kernel 5: out = sampled @ W_out + b_out, scattered back to NCHW per level.
// d_out = [ (8,128,64,64) | (8,128,32,32) | (8,128,16,16) ] fp32, flat.
// Per lane: two aligned float4 stores (s-base is 8-aligned within a level).
// ---------------------------------------------------------------------------
__global__ void gemm_out_kernel(const _Float16* __restrict__ sampled_h,
                                const _Float16* __restrict__ wout_frag,
                                const float* __restrict__ b_out,
                                float* __restrict__ out) {
    int qt   = blockIdx.x;
    int b    = blockIdx.y;
    int wave = threadIdx.x >> 5;
    int lane = threadIdx.x & 31;
    size_t rowbase = (size_t)b * LQ + (size_t)qt * 16;
    const _Float16* xrow = sampled_h + rowbase * D_MODEL;
    v16h a[KT];
#pragma unroll
    for (int kt = 0; kt < KT; ++kt) a[kt] = load_a_frag(xrow, lane, kt * 32);

    // whole 16-query tile lies in one level (boundaries 4096/5120 are 16-aligned)
    int q0 = qt * 16;
    int lvl, start, Wl;
    level_info(q0, lvl, start, Wl);
    int hw = Wl * Wl;
    size_t obase = (lvl == 0) ? 0
                 : (lvl == 1) ? (size_t)BATCH * D_MODEL * 4096
                              : (size_t)BATCH * D_MODEL * 4096 + (size_t)BATCH * D_MODEL * 1024;

    const int NT = D_MODEL / 16;   // 8
#pragma unroll
    for (int i = 0; i < 2; ++i) {
        int nt = wave * 2 + i;
        v8f acc = wmma_tile(a, wout_frag, nt, NT, lane);
        int n = nt * 16 + (lane & 15);   // output channel c
        float bias = b_out[n];
        int sbase = q0 - start + ((lane & 16) ? 8 : 0);   // 8-aligned
        float* dst = out + obase + ((size_t)b * D_MODEL + n) * hw + sbase;
        float4 lo = make_float4(acc[0] + bias, acc[1] + bias, acc[2] + bias, acc[3] + bias);
        float4 hi = make_float4(acc[4] + bias, acc[5] + bias, acc[6] + bias, acc[7] + bias);
        *(float4*)(dst)     = lo;
        *(float4*)(dst + 4) = hi;
    }
}

// ---------------------------------------------------------------------------
// Launch
// ---------------------------------------------------------------------------
extern "C" void kernel_launch(void* const* d_in, const int* in_sizes, int n_in,
                              void* d_out, int out_size, void* d_ws, size_t ws_size,
                              hipStream_t stream) {
    (void)in_sizes; (void)n_in; (void)out_size; (void)ws_size;
    const float* fm1    = (const float*)d_in[0];
    const float* fm2    = (const float*)d_in[1];
    const float* fm3    = (const float*)d_in[2];
    const float* W_off  = (const float*)d_in[3];
    const float* b_off  = (const float*)d_in[4];
    const float* W_attn = (const float*)d_in[5];
    const float* b_attn = (const float*)d_in[6];
    const float* W_val  = (const float*)d_in[7];
    const float* b_val  = (const float*)d_in[8];
    const float* W_out  = (const float*)d_in[9];
    const float* b_out  = (const float*)d_in[10];
    float* out = (float*)d_out;

    // workspace layout (256B aligned): ~33.3 MB total
    char*  ws  = (char*)d_ws;
    size_t off = 0;
    auto alloc = [&](size_t bytes) -> void* {
        void* p = ws + off;
        off += (bytes + 255) & ~(size_t)255;
        return p;
    };
    const size_t TOK = (size_t)BATCH * LQ;                      // 43008
    _Float16* xh        = (_Float16*)alloc(TOK * D_MODEL * sizeof(_Float16)); // 11.0 MB
    _Float16* wval_frag = (_Float16*)alloc((size_t)128 * 128 * sizeof(_Float16));
    _Float16* woff_frag = (_Float16*)alloc((size_t)128 * 384 * sizeof(_Float16));
    _Float16* wattn_frag= (_Float16*)alloc((size_t)128 * 192 * sizeof(_Float16));
    _Float16* wout_frag = (_Float16*)alloc((size_t)128 * 128 * sizeof(_Float16));
    _Float16* value     = (_Float16*)alloc(TOK * D_MODEL * sizeof(_Float16)); // 11.0 MB
    _Float16* sampled_h = (_Float16*)alloc(TOK * D_MODEL * sizeof(_Float16)); // 11.0 MB

    // 1) pack activations to fp16 token-major (LDS transpose, coalesced both sides)
    pack_x_kernel<<<dim3(168, BATCH), 256, 0, stream>>>(fm1, fm2, fm3, xh);
    // 2) swizzle weights into WMMA B-fragment order
    {
        int total = 128 * (128 + 384 + 192 + 128);   // 106496
        pack_w_kernel<<<(total + 255) / 256, 256, 0, stream>>>(
            W_val, W_off, W_attn, W_out, wval_frag, woff_frag, wattn_frag, wout_frag);
    }
    dim3 grid(LQ / 16, BATCH);   // (336, 8) 16-query tiles
    // 3) value projection (fp16 out for cheap gathers)
    gemm_value_kernel<<<grid, 128, 0, stream>>>(xh, wval_frag, b_val, value);
    // 4) fused offsets/attn GEMM + softmax + bilinear sampling
    msda_sample_kernel<<<grid, 128, 0, stream>>>(xh, woff_frag, wattn_frag,
                                                 b_off, b_attn, value, sampled_h);
    // 5) output projection + vectorized scatter to NCHW per level
    gemm_out_kernel<<<grid, 128, 0, stream>>>(sampled_h, wout_frag, b_out, out);
}
